// PostProcessCocoGrounding_60739427500178
// MI455X (gfx1250) — matrix operand order repaired and checked by reading, not compiled, not run on hardware
//
#include <hip/hip_runtime.h>
#include <hip/hip_bf16.h>

typedef __attribute__((ext_vector_type(16))) _Float16 v16h;
typedef __attribute__((ext_vector_type(8)))  _Float16 v8h;
typedef __attribute__((ext_vector_type(8)))  float    v8f;
typedef __attribute__((ext_vector_type(4)))  float    v4f;

#define B_    256
#define Q_    900
#define T_    256
#define C_    91
#define CP_   96
#define K_SEL 300
#define NQC   (Q_ * C_)          // 81900 candidates per batch

// ---------------------------------------------------------------------------
// Stage 1:  prob[b,q,c] = sum_t sigmoid(logits[b,q,t]) * pos_map[c,t]
// Batched GEMM on the WMMA pipe: f16 inputs (sigmoid in [0,1]; accumulate
// in f32), v_wmma_f32_16x16x32_f16 tiles. Memory-bound on the 236 MB logits
// stream; sigmoid uses v_exp_f32 + v_rcp_f32 (approx rcp is exact enough for
// an f16 destination) to stay off the critical path.
// grid = (15 q-tiles, 256 batches), block = 256 threads = 8 wave32s.
// Each block: 64 q-rows x 96 classes = 24 tiles of 16x16, 3 tiles per wave.
// ---------------------------------------------------------------------------
__global__ __launch_bounds__(256)
void pp_gemm_wmma(const float* __restrict__ logits,
                  const float* __restrict__ pos_map,
                  float* __restrict__ prob) {
  // A tile: 64 rows x 32 K, f16, row stride 40 halves (80B: 16B-aligned, bank-spread)
  __shared__ _Float16 ash[64][40];
  // pos_map^T as [class][t], full K=256 resident, stride 264 halves (528B)
  __shared__ _Float16 bsh[CP_][264];

  const int tid  = threadIdx.x;
  const int lane = tid & 31;
  const int wave = tid >> 5;
  const int q0   = blockIdx.x * 64;       // 0..896
  const int b    = blockIdx.y;            // 0..255

  // Stage pos_map into LDS as f16 (classes 91..95 zero-padded)
  for (int i = tid; i < CP_ * T_; i += 256) {
    const int c = i >> 8, t = i & 255;
    const float v = (c < C_) ? pos_map[c * T_ + t] : 0.0f;
    bsh[c][t] = (_Float16)v;
  }

  v8f acc[3];
  #pragma unroll
  for (int j = 0; j < 3; ++j) { v8f z = {}; acc[j] = z; }

  const float* lg = logits + (size_t)b * Q_ * T_;

  const int kbA = (lane >> 4) * 8;        // A fragment K base: 0 or 8
  const int kbB = (lane >> 4) * 8;        // B fragment K base: 0 or 8
  const int mn  = lane & 15;

  for (int kk = 0; kk < 8; ++kk) {        // K = 8 x 32 = 256
    __syncthreads();
    // ---- stage A: 64x32 logits -> sigmoid -> f16 LDS ----
    {
      const int row = tid >> 2;           // 0..63
      const int kc  = (tid & 3) * 8;      // 0,8,16,24
      const int q   = q0 + row;
      _Float16 h[8];
      if (q < Q_) {
        const float* src = lg + (size_t)q * T_ + kk * 32 + kc;
        const v4f x0 = *(const v4f*)(src);
        const v4f x1 = *(const v4f*)(src + 4);
        const float xs[8] = {x0.x, x0.y, x0.z, x0.w, x1.x, x1.y, x1.z, x1.w};
        #pragma unroll
        for (int i = 0; i < 8; ++i) {
          // sigmoid via v_exp_f32 + v_rcp_f32: ~1 ulp f32 error, invisible
          // after the f16 truncation; avoids the IEEE div fixup sequence.
          const float s = __builtin_amdgcn_rcpf(1.0f + __expf(-xs[i]));
          h[i] = (_Float16)s;
        }
        if (kk < 7) __builtin_prefetch(src + 32, 0, 0);   // global_prefetch_b8
      } else {
        #pragma unroll
        for (int i = 0; i < 8; ++i) h[i] = (_Float16)0.0f;
      }
      #pragma unroll
      for (int i = 0; i < 8; ++i) ash[row][kc + i] = h[i];
    }
    __syncthreads();

    // ---- 3 WMMA tiles per wave ----
    #pragma unroll
    for (int j = 0; j < 3; ++j) {
      const int tidx = wave + j * 8;      // 0..23
      const int mi = tidx / 6, ni = tidx % 6;

      // A fragment (16x32 f16): lane row M=mn, halves: K = kbA+0..7, kbA+16..23
      const v8h alo = *(const v8h*)&ash[mi * 16 + mn][kbA];
      const v8h ahi = *(const v8h*)&ash[mi * 16 + mn][kbA + 16];
      v16h a;
      #pragma unroll
      for (int i = 0; i < 8; ++i) { a[i] = alo[i]; a[i + 8] = ahi[i]; }

      // B fragment (32x16 f16): lane col N=mn, halves: K = kbB+0..7, kbB+16..23
      const v8h blo = *(const v8h*)&bsh[ni * 16 + mn][kk * 32 + kbB];
      const v8h bhi = *(const v8h*)&bsh[ni * 16 + mn][kk * 32 + kbB + 16];
      v16h bb;
      #pragma unroll
      for (int i = 0; i < 8; ++i) { bb[i] = blo[i]; bb[i + 8] = bhi[i]; }

      acc[j] = __builtin_amdgcn_wmma_f32_16x16x32_f16(
          false, a, false, bb, (short)0, acc[j], false, false);
    }
  }

  // ---- write back: D VGPR r holds M = r + 8*(lane>=16), N = lane&15 ----
  const int moff = (lane >> 4) * 8;
  #pragma unroll
  for (int j = 0; j < 3; ++j) {
    const int tidx = wave + j * 8;
    const int mi = tidx / 6, ni = tidx % 6;
    const int c = ni * 16 + mn;
    #pragma unroll
    for (int r = 0; r < 8; ++r) {
      const int q = q0 + mi * 16 + moff + r;
      if (q < Q_ && c < C_)
        prob[((size_t)b * Q_ + q) * C_ + c] = acc[j][r];
    }
  }
}

// ---------------------------------------------------------------------------
// Stage 2: per-batch exact top-300 via 3-pass radix refinement on float bits
// (scores >= 0 so the uint bit pattern is order-preserving), then 512-wide
// bitonic sort, then fused box gather / cxcywh->xyxy / scale epilogue.
// grid = 256 blocks (one per batch), 256 threads.
// ---------------------------------------------------------------------------
__global__ __launch_bounds__(256)
void pp_topk(const float* __restrict__ prob,
             const float* __restrict__ boxes_in,
             const int* __restrict__ tsizes,
             float* __restrict__ out) {
  __shared__ unsigned hist[4096];
  __shared__ float    sval[512];
  __shared__ int      sidx[512];
  __shared__ unsigned sThr, sAbove, sCnt, sCntEq;

  const int b   = blockIdx.x;
  const int tid = threadIdx.x;
  const float* p = prob + (size_t)b * NQC;

  // ---- pass 1: bits[31:22] ----
  for (int i = tid; i < 1024; i += 256) hist[i] = 0;
  __syncthreads();
  for (int i = tid; i < NQC; i += 256)
    atomicAdd(&hist[__float_as_uint(p[i]) >> 22], 1u);
  __syncthreads();
  if (tid == 0) {
    unsigned cum = 0, t = 0;
    for (int i = 1023; i >= 0; --i) {
      if (cum + hist[i] >= K_SEL) { t = (unsigned)i; break; }
      cum += hist[i];
    }
    sThr = t; sAbove = cum;
  }
  __syncthreads();
  const unsigned T1 = sThr; const unsigned above1 = sAbove;
  __syncthreads();

  // ---- pass 2: bits[21:12] among prefix==T1 ----
  for (int i = tid; i < 1024; i += 256) hist[i] = 0;
  __syncthreads();
  for (int i = tid; i < NQC; i += 256) {
    const unsigned k = __float_as_uint(p[i]);
    if ((k >> 22) == T1) atomicAdd(&hist[(k >> 12) & 1023u], 1u);
  }
  __syncthreads();
  if (tid == 0) {
    unsigned cum = above1, t = 0;
    for (int i = 1023; i >= 0; --i) {
      if (cum + hist[i] >= K_SEL) { t = (unsigned)i; break; }
      cum += hist[i];
    }
    sThr = t; sAbove = cum;
  }
  __syncthreads();
  const unsigned T2 = sThr; const unsigned above2 = sAbove;
  const unsigned pref = (T1 << 10) | T2;
  __syncthreads();

  // ---- pass 3: bits[11:0] among prefix==(T1,T2) ----
  for (int i = tid; i < 4096; i += 256) hist[i] = 0;
  __syncthreads();
  for (int i = tid; i < NQC; i += 256) {
    const unsigned k = __float_as_uint(p[i]);
    if ((k >> 12) == pref) atomicAdd(&hist[k & 0xFFFu], 1u);
  }
  __syncthreads();
  if (tid == 0) {
    unsigned cum = above2, t = 0;
    for (int i = 4095; i >= 0; --i) {
      if (cum + hist[i] >= K_SEL) { t = (unsigned)i; break; }
      cum += hist[i];
    }
    sThr = t; sAbove = cum;
    sCnt = 0; sCntEq = 0;
  }
  __syncthreads();
  const unsigned thrKey = (pref << 12) | sThr;   // 300th-largest key
  const unsigned above  = sAbove;                // count strictly above (<300)
  __syncthreads();

  // ---- collect strictly-above, then fill ties at the threshold ----
  for (int i = tid; i < NQC; i += 256) {
    const float v = p[i];
    if (__float_as_uint(v) > thrKey) {
      const unsigned pos = atomicAdd(&sCnt, 1u);
      sval[pos] = v; sidx[pos] = i;
    }
  }
  __syncthreads();
  for (int i = tid; i < NQC; i += 256) {
    const float v = p[i];
    if (__float_as_uint(v) == thrKey) {
      const unsigned pos = atomicAdd(&sCntEq, 1u);
      if (above + pos < K_SEL) { sval[above + pos] = v; sidx[above + pos] = i; }
    }
  }
  __syncthreads();
  for (int i = K_SEL + tid; i < 512; i += 256) { sval[i] = -1.0f; sidx[i] = 0x7FFFFFFF - i; }
  __syncthreads();

  // ---- bitonic sort, descending by value, ascending index on ties ----
  for (unsigned kk = 2; kk <= 512; kk <<= 1) {
    for (unsigned j = kk >> 1; j > 0; j >>= 1) {
      for (unsigned i = tid; i < 512; i += 256) {
        const unsigned ixj = i ^ j;
        if (ixj > i) {
          const float v0 = sval[i], v1 = sval[ixj];
          const int   i0 = sidx[i], i1 = sidx[ixj];
          const bool pre  = (v0 > v1) || (v0 == v1 && i0 < i1);
          const bool desc = ((i & kk) == 0);
          if (desc ? !pre : pre) {
            sval[i] = v1; sval[ixj] = v0;
            sidx[i] = i1; sidx[ixj] = i0;
          }
        }
      }
      __syncthreads();
    }
  }

  // ---- epilogue: scores / labels / scaled xyxy boxes ----
  const float wIm = (float)tsizes[b * 2 + 1];
  const float hIm = (float)tsizes[b * 2 + 0];
  for (int k = tid; k < K_SEL; k += 256) {
    const float v = sval[k];
    const int idx = sidx[k];
    const int q = idx / C_;
    const int c = idx - q * C_;
    const float* bx = boxes_in + ((size_t)b * Q_ + q) * 4;
    const float cx = bx[0], cy = bx[1], bw = bx[2], bh = bx[3];
    out[(size_t)b * K_SEL + k]                 = v;           // scores
    out[(size_t)B_ * K_SEL + b * K_SEL + k]    = (float)c;    // labels
    float* ob = out + 2 * (size_t)B_ * K_SEL + ((size_t)(b * K_SEL + k)) * 4;
    ob[0] = (cx - 0.5f * bw) * wIm;
    ob[1] = (cy - 0.5f * bh) * hIm;
    ob[2] = (cx + 0.5f * bw) * wIm;
    ob[3] = (cy + 0.5f * bh) * hIm;
  }
}

extern "C" void kernel_launch(void* const* d_in, const int* in_sizes, int n_in,
                              void* d_out, int out_size, void* d_ws, size_t ws_size,
                              hipStream_t stream) {
  (void)in_sizes; (void)n_in; (void)out_size; (void)ws_size;
  const float* logits = (const float*)d_in[0];   // [256,900,256]
  const float* boxes  = (const float*)d_in[1];   // [256,900,4]
  const float* posmap = (const float*)d_in[2];   // [91,256]
  const int*   tsz    = (const int*)d_in[3];     // [256,2]
  float* prob = (float*)d_ws;                    // [256,900,91] fp32 = 83.9 MB

  dim3 g1((Q_ + 63) / 64, B_);                   // 15 x 256 blocks
  pp_gemm_wmma<<<g1, 256, 0, stream>>>(logits, posmap, prob);
  pp_topk<<<B_, 256, 0, stream>>>(prob, boxes, tsz, (float*)d_out);
}
